// Overlapped_Window_Cross_Level_Semantic_Guidance_10746008175230
// MI455X (gfx1250) — matrix operand order repaired and checked by reading, not compiled, not run on hardware
//
#include <hip/hip_runtime.h>
#include <hip/hip_bf16.h>

typedef __attribute__((ext_vector_type(16))) __bf16         v16bf;
typedef __attribute__((ext_vector_type(16))) unsigned short v16u16;
typedef __attribute__((ext_vector_type(8)))  unsigned short v8u16;
typedef __attribute__((ext_vector_type(8)))  float          v8f;

// ---------------- problem constants ----------------
constexpr int BN  = 8;     // batch
constexpr int CH  = 256;   // high channels
constexpr int CB  = 128;   // base channels
constexpr int HH  = 64;    // high H=W
constexpr int HL  = 128;   // low H=W
constexpr int PH  = HH * HH;   // 4096 pixels (high/hs)
constexpr int PL  = HL * HL;   // 16384 pixels (low)
constexpr int NH1 = 15;    // windows/dim, ws=8  (hs)  == wsq=16 (low)
constexpr int NH2 = 7;     // windows/dim, ws=16 (hs)  == wsq=32 (low)

// ---------------- workspace layout (bytes) ----------------
constexpr size_t OFF_CONVF = 0;                               // u16[8*8*32*16] fragment-ready
constexpr size_t OFF_WQF   = OFF_CONVF + 256 * 128 * 2;
constexpr size_t OFF_WKF   = OFF_WQF   + 128 * 128 * 2;
constexpr size_t OFF_WVF   = OFF_WKF   + 128 * 128 * 2;
constexpr size_t OFF_WOF   = OFF_WVF   + 128 * 128 * 2;
constexpr size_t OFF_Q     = OFF_WOF   + 128 * 128 * 2;       // u16[BN*PL*CB]  pixel-major [p][c]
constexpr size_t OFF_K     = OFF_Q  + (size_t)BN * PL * CB * 2; // u16 channel-major [b][c][p]
constexpr size_t OFF_V     = OFF_K  + (size_t)BN * PH * CB * 2; // u16 channel-major [b][c][p]
constexpr size_t OFF_KV8   = OFF_V  + (size_t)BN * PH * CB * 2; // f32[1800][4][32][32]
constexpr size_t OFF_KM8   = OFF_KV8  + (size_t)BN * NH1 * NH1 * 4096 * 4;
constexpr size_t OFF_KV16  = OFF_KM8  + (size_t)BN * NH1 * NH1 * 128 * 4;
constexpr size_t OFF_KM16  = OFF_KV16 + (size_t)BN * NH2 * NH2 * 4096 * 4;

// ---------------- helpers ----------------
__device__ inline unsigned short f2bf(float f) {
  unsigned u = __float_as_uint(f);
  u += 0x7fffu + ((u >> 16) & 1u);
  return (unsigned short)(u >> 16);
}
__device__ inline float bf2f(unsigned short h) {
  return __uint_as_float(((unsigned)h) << 16);
}
__device__ inline v8f wmma_bf16(v16u16 a, v16u16 b, v8f c) {
  return __builtin_amdgcn_wmma_f32_16x16x32_bf16(
      false, __builtin_bit_cast(v16bf, a),
      false, __builtin_bit_cast(v16bf, b),
      (short)0, c, false, false);
}
__device__ inline v16u16 cat16(v8u16 lo, v8u16 hi) {
  return __builtin_shufflevector(lo, hi, 0, 1, 2, 3, 4, 5, 6, 7,
                                 8, 9, 10, 11, 12, 13, 14, 15);
}
// A fragment (16x32 bf16): lane half owns K run {h*16..h*16+15}
__device__ inline v16u16 loadA(const unsigned short* row, int half) {
  v8u16 lo = *(const v8u16*)(row + half * 16);
  v8u16 hi = *(const v8u16*)(row + half * 16 + 8);
  return cat16(lo, hi);
}

// -------- CDNA5 async global->LDS (ASYNCcnt) --------
typedef __attribute__((address_space(3))) unsigned char lds_b;
__device__ inline unsigned ldsoff(const void* p) {
  return (unsigned)(unsigned long long)(lds_b*)p;   // LDS byte offset
}
__device__ inline void async_ld_b64(void* lds, const void* g) {
  asm volatile("global_load_async_to_lds_b64 %0, %1, off"
               :: "v"(ldsoff(lds)), "v"((unsigned long long)g) : "memory");
}
__device__ inline void async_ld_b128(void* lds, const void* g) {
  asm volatile("global_load_async_to_lds_b128 %0, %1, off"
               :: "v"(ldsoff(lds)), "v"((unsigned long long)g) : "memory");
}
__device__ inline void wait_async0() {
  asm volatile("s_wait_asynccnt 0x0" ::: "memory");
}

__device__ inline int cov1d(int x, int wsq, int st, int n) {
  int imax = x / st; if (imax > n - 1) imax = n - 1;
  int imin = (x >= wsq) ? ((x - wsq) / st + 1) : 0;
  int c = imax - imin + 1;
  return c > 0 ? c : 0;
}

// ---------------- kernel 0: weights -> bf16 fragment-ready ----------------
// frag flat index i = ((kc*8 + ntile)*32 + lane)*16 + j
// value = W[n][k],  n = ntile*16 + (lane&15),  k = kc*32 + (lane>>4)*16 + j
__global__ __launch_bounds__(256) void k_prep(const float* conv_w, const float* wq,
                                              const float* wk, const float* wv,
                                              const float* wo, unsigned char* wsb) {
  int i = blockIdx.x * 256 + threadIdx.x;
  unsigned short* convF = (unsigned short*)(wsb + OFF_CONVF);
  unsigned short* wqF   = (unsigned short*)(wsb + OFF_WQF);
  unsigned short* wkF   = (unsigned short*)(wsb + OFF_WKF);
  unsigned short* wvF   = (unsigned short*)(wsb + OFF_WVF);
  unsigned short* woF   = (unsigned short*)(wsb + OFF_WOF);
  int j = i & 15, lane = (i >> 4) & 31, nt = (i >> 9) & 7;
  int n = nt * 16 + (lane & 15);
  int kk = ((lane >> 4) * 16) + j;
  if (i < 256 * 128) {                       // conv: K = 256, kc = i>>12 (0..7)
    int k = ((i >> 12) & 7) * 32 + kk;
    convF[i] = f2bf(conv_w[n * 256 + k]);
  }
  if (i < 128 * 128) {                       // CB x CB: K = 128, kc = i>>12 (0..3)
    int k = ((i >> 12) & 3) * 32 + kk;
    wqF[i] = f2bf(wq[n * 128 + k]);
    wkF[i] = f2bf(wk[n * 128 + k]);
    wvF[i] = f2bf(wv[n * 128 + k]);
    woF[i] = f2bf(wo[n * 128 + k]);
  }
}

__global__ __launch_bounds__(256) void k_zero(float* p, int n) {
  int i = blockIdx.x * 256 + threadIdx.x;
  if (i < n) p[i] = 0.f;
}

// ---------------- kernel 1: conv1x1 + LN + K/V projections (hs image) ----------------
__global__ __launch_bounds__(256) void k_hs(const float* high, const float* conv_b,
                                            const float* ln_g, const float* ln_b,
                                            const float* bk, const float* bv,
                                            unsigned char* wsb) {
  __shared__ unsigned short ain[16][256];   // A tile (pixel-row, channel-contig)
  __shared__ float hbuf[16][128];
  __shared__ unsigned short aln[16][128];
  __shared__ float mu[16], rs[16];
  const unsigned short* convF = (const unsigned short*)(wsb + OFF_CONVF);
  const unsigned short* wkF   = (const unsigned short*)(wsb + OFF_WKF);
  const unsigned short* wvF   = (const unsigned short*)(wsb + OFF_WVF);
  unsigned short* kf = (unsigned short*)(wsb + OFF_K);
  unsigned short* vf = (unsigned short*)(wsb + OFF_V);

  const int tid = threadIdx.x, lane = tid & 31, wave = tid >> 5;
  const int half = lane >> 4, ml = lane & 15;
  const int pix0 = blockIdx.x * 16, b = pix0 / PH, p0 = pix0 % PH;

  for (int s = tid; s < 256 * 4; s += 256) {  // f32->bf16 transpose staging
    int c = s >> 2, seg = s & 3;
    float4 f = *(const float4*)&high[(size_t)(b * CH + c) * PH + p0 + seg * 4];
    ain[seg * 4 + 0][c] = f2bf(f.x);
    ain[seg * 4 + 1][c] = f2bf(f.y);
    ain[seg * 4 + 2][c] = f2bf(f.z);
    ain[seg * 4 + 3][c] = f2bf(f.w);
  }
  __syncthreads();

  // conv: X[16,256] @ W^T[256,128]; B fragments straight from global (L2-hot)
  v8f acc = {0.f, 0.f, 0.f, 0.f, 0.f, 0.f, 0.f, 0.f};
  for (int kc = 0; kc < 8; ++kc) {
    v16u16 a = loadA(&ain[ml][kc * 32], half);
    v16u16 bb = *(const v16u16*)(convF + ((kc * 8 + wave) * 32 + lane) * 16);
    acc = wmma_bf16(a, bb, acc);
  }
  {
    int c = wave * 16 + ml;
    float cb = conv_b[c];
#pragma unroll
    for (int r = 0; r < 8; ++r) hbuf[half * 8 + r][c] = acc[r] + cb;
  }
  __syncthreads();

  if (tid < 16) {  // per-pixel LN stats over 128 channels
    float s = 0.f, s2 = 0.f;
    for (int c = 0; c < 128; ++c) { float x = hbuf[tid][c]; s += x; s2 += x * x; }
    float m = s * (1.f / 128.f);
    float v = s2 * (1.f / 128.f) - m * m;
    mu[tid] = m; rs[tid] = rsqrtf(v + 1e-5f);
  }
  __syncthreads();
  for (int i = tid; i < 16 * 128; i += 256) {
    int m = i >> 7, c = i & 127;
    aln[m][c] = f2bf((hbuf[m][c] - mu[m]) * rs[m] * ln_g[c] + ln_b[c]);
  }
  __syncthreads();

  // K and V projections (pass 0: k = elu(x)+1 ; pass 1: v); channel-major output
  for (int pass = 0; pass < 2; ++pass) {
    const unsigned short* WF = (pass == 0) ? wkF : wvF;
    const float* bias = (pass == 0) ? bk : bv;
    unsigned short* dst = (pass == 0) ? kf : vf;
    v8f a2 = {0.f, 0.f, 0.f, 0.f, 0.f, 0.f, 0.f, 0.f};
    for (int kc = 0; kc < 4; ++kc) {
      v16u16 a = loadA(&aln[ml][kc * 32], half);
      v16u16 bb = *(const v16u16*)(WF + ((kc * 8 + wave) * 32 + lane) * 16);
      a2 = wmma_bf16(a, bb, a2);
    }
    int c = wave * 16 + ml;
    float bs = bias[c];
    v8u16 px;
#pragma unroll
    for (int r = 0; r < 8; ++r) {
      float val = a2[r] + bs;
      if (pass == 0) val = (val > 0.f) ? (val + 1.f) : __expf(val);
      px[r] = f2bf(val);
    }
    *(v8u16*)&dst[(size_t)(b * CB + c) * PH + p0 + half * 8] = px;  // 8 contig pixels
  }
}

// ---------------- kernel 2: LN + Q projection (low image) ----------------
__global__ __launch_bounds__(256) void k_q(const float* low, const float* ln_g,
                                           const float* ln_b, const float* bq,
                                           unsigned char* wsb) {
  __shared__ float xin[16][128];
  __shared__ unsigned short aln[16][128];
  __shared__ float mu[16], rs[16];
  const unsigned short* wqF = (const unsigned short*)(wsb + OFF_WQF);
  unsigned short* qf = (unsigned short*)(wsb + OFF_Q);

  const int tid = threadIdx.x, lane = tid & 31, wave = tid >> 5;
  const int half = lane >> 4, ml = lane & 15;
  const int pix0 = blockIdx.x * 16, b = pix0 / PL, p0 = pix0 % PL;

  for (int s = tid; s < 128 * 4; s += 256) {
    int c = s >> 2, seg = s & 3;
    float4 f = *(const float4*)&low[(size_t)(b * CB + c) * PL + p0 + seg * 4];
    xin[seg * 4 + 0][c] = f.x;
    xin[seg * 4 + 1][c] = f.y;
    xin[seg * 4 + 2][c] = f.z;
    xin[seg * 4 + 3][c] = f.w;
  }
  __syncthreads();
  if (tid < 16) {
    float s = 0.f, s2 = 0.f;
    for (int c = 0; c < 128; ++c) { float x = xin[tid][c]; s += x; s2 += x * x; }
    float m = s * (1.f / 128.f);
    float v = s2 * (1.f / 128.f) - m * m;
    mu[tid] = m; rs[tid] = rsqrtf(v + 1e-5f);
  }
  __syncthreads();
  for (int i = tid; i < 16 * 128; i += 256) {
    int m = i >> 7, c = i & 127;
    aln[m][c] = f2bf((xin[m][c] - mu[m]) * rs[m] * ln_g[c] + ln_b[c]);
  }
  __syncthreads();

  v8f acc = {0.f, 0.f, 0.f, 0.f, 0.f, 0.f, 0.f, 0.f};
  for (int kc = 0; kc < 4; ++kc) {
    v16u16 a = loadA(&aln[ml][kc * 32], half);
    v16u16 bb = *(const v16u16*)(wqF + ((kc * 8 + wave) * 32 + lane) * 16);
    acc = wmma_bf16(a, bb, acc);
  }
  int c = wave * 16 + ml;
  float bqv = bq[c];
#pragma unroll
  for (int r = 0; r < 8; ++r) {
    int m = half * 8 + r;
    float val = acc[r] + bqv;
    val = (val > 0.f) ? (val + 1.f) : __expf(val);   // elu(x)+1
    qf[(size_t)(b * PL + p0 + m) * CB + c] = f2bf(val);  // pixel-major [p][c]
  }
}

// ---------------- kernel 3: per-window kv = k^T v / ws and kmean ----------------
__global__ __launch_bounds__(256) void k_kv(unsigned char* wsb, int ws, int nh,
                                            unsigned long long kvoff,
                                            unsigned long long kmoff) {
  __shared__ unsigned short kbufT[128][32];   // [channel][window-pixel]
  __shared__ unsigned short vbufT[128][32];
  __shared__ float kmb[128];
  const unsigned short* kf = (const unsigned short*)(wsb + OFF_K);
  const unsigned short* vf = (const unsigned short*)(wsb + OFF_V);
  float* kvp = (float*)(wsb + kvoff);
  float* kmp = (float*)(wsb + kmoff);

  const int tid = threadIdx.x, lane = tid & 31, wave = tid >> 5;
  const int half = lane >> 4, ml = lane & 15;
  const int w = blockIdx.x;
  const int wx = w % nh, wy = (w / nh) % nh, b = w / (nh * nh);
  const int st = ws / 2;
  const int head = wave >> 1, mtile = wave & 1;
  const int ca = head * 32 + mtile * 16;

  if (tid < 128) kmb[tid] = 0.f;
  __syncthreads();

  v8f acc0 = {0.f, 0.f, 0.f, 0.f, 0.f, 0.f, 0.f, 0.f};
  v8f acc1 = {0.f, 0.f, 0.f, 0.f, 0.f, 0.f, 0.f, 0.f};
  const int chunks = (ws * ws) / 32;
  for (int ch = 0; ch < chunks; ++ch) {
    // stage 32 window pixels channel-major via async global->LDS (8B segments)
    for (int s = tid; s < 128 * 8; s += 256) {
      int c = s >> 3, seg = s & 7;
      int nn0 = seg * 4;
      int n = ch * 32 + nn0;
      int y = wy * st + n / ws, x = wx * st + n % ws;   // x multiple of 4
      size_t gp = (size_t)(b * CB + c) * PH + y * HH + x;
      async_ld_b64(&kbufT[c][nn0], &kf[gp]);
      async_ld_b64(&vbufT[c][nn0], &vf[gp]);
    }
    if (ch + 1 < chunks) {       // hint next chunk toward L2 (global_prefetch_b8)
      int c = tid >> 1, seg = (tid & 1) * 16;
      int n = (ch + 1) * 32 + seg;
      int y = wy * st + n / ws, x = wx * st + n % ws;
      size_t gp = (size_t)(b * CB + c) * PH + y * HH + x;
      __builtin_prefetch(&kf[gp], 0, 1);
    }
    wait_async0();
    __syncthreads();
    if (tid < 128) {
      float s = 0.f;
      for (int nn = 0; nn < 32; ++nn) s += bf2f(kbufT[tid][nn]);
      kmb[tid] += s;
    }
    v16u16 a  = loadA(&kbufT[ca + ml][0], half);                       // k^T tile
    v16u16 b0 = *(const v16u16*)&vbufT[head * 32 + ml][half * 16];     // 32B run
    v16u16 b1 = *(const v16u16*)&vbufT[head * 32 + 16 + ml][half * 16];
    acc0 = wmma_bf16(a, b0, acc0);
    acc1 = wmma_bf16(a, b1, acc1);
    __syncthreads();
  }
  const float inv = 1.0f / (float)ws;
  float* out = kvp + (size_t)w * 4096 + head * 1024;
#pragma unroll
  for (int r = 0; r < 8; ++r) {
    int d = mtile * 16 + half * 8 + r;
    out[d * 32 + ml] = acc0[r] * inv;
    out[d * 32 + 16 + ml] = acc1[r] * inv;
  }
  if (tid < 128) kmp[(size_t)w * 128 + tid] = kmb[tid] / (float)(ws * ws);
}

// ---------------- kernel 4: out = (q @ kv) * z * ww, folded (scatter-add) ----------------
__global__ __launch_bounds__(256) void k_att(unsigned char* wsb, const float* wwv, int wwidx,
                                             float* attn_acc, int ws, int nh, int tiles,
                                             unsigned long long kvoff,
                                             unsigned long long kmoff) {
  __shared__ unsigned short qbuf[16][128];
  __shared__ unsigned short kvf[8][32][16];  // fragment-ready: [wave][lane][j]
  __shared__ float kmb[128];
  __shared__ float zbuf[16][4];
  const unsigned short* qf = (const unsigned short*)(wsb + OFF_Q);
  const float* kvp = (const float*)(wsb + kvoff);
  const float* kmp = (const float*)(wsb + kmoff);

  const int tid = threadIdx.x, lane = tid & 31, wave = tid >> 5;
  const int half = lane >> 4, ml = lane & 15;
  const int w = blockIdx.x / tiles, tile = blockIdx.x % tiles;
  const int wx = w % nh, wy = (w / nh) % nh, b = w / (nh * nh);
  const int wsq = 2 * ws, stl = ws;  // low-image window & stride

  // q tile: async 16B global->LDS moves (q is pixel-major, channel runs contiguous)
  {
    int m = tid >> 4, seg = tid & 15;   // 256 segments of 8 channels
    int n = tile * 16 + m;
    int y = wy * stl + n / wsq, x = wx * stl + n % wsq;
    size_t p = (size_t)(b * PL) + y * HL + x;
    async_ld_b128(&qbuf[m][seg * 8], &qf[p * CB + seg * 8]);
  }
  // kv repack into fragment order (f32 -> bf16, so scalar gather):
  // t = head*2+ntile; value = kv[head][d = (l>>4)*16 + j][e = ntile*16 + (l&15)]
  for (int i = tid; i < 4096; i += 256) {
    int j = i & 15, l = (i >> 4) & 31, t = i >> 9;
    int hd = t >> 1, nt = t & 1;
    int d = (l >> 4) * 16 + j, e = nt * 16 + (l & 15);
    ((unsigned short*)kvf)[i] = f2bf(kvp[(size_t)w * 4096 + hd * 1024 + d * 32 + e]);
  }
  if (tid < 128) kmb[tid] = kmp[(size_t)w * 128 + tid];
  wait_async0();
  __syncthreads();

  if (tid < 64) {   // z = 1/(q . kmean + eps) per (pixel, head)
    int m = tid >> 2, h = tid & 3;
    float s = 1e-6f;
    for (int d = 0; d < 32; ++d) s += bf2f(qbuf[m][h * 32 + d]) * kmb[h * 32 + d];
    zbuf[m][h] = 1.0f / s;
  }
  __syncthreads();

  const int head = wave >> 1, ntile = wave & 1;
  v16u16 a  = loadA(&qbuf[ml][head * 32], half);
  v16u16 bb = *(const v16u16*)&kvf[wave][lane][0];
  v8f acc = {0.f, 0.f, 0.f, 0.f, 0.f, 0.f, 0.f, 0.f};
  acc = wmma_bf16(a, bb, acc);

  const float ww = wwv[wwidx];
  const int c = head * 32 + ntile * 16 + ml;
#pragma unroll
  for (int r = 0; r < 8; ++r) {
    int m = half * 8 + r;
    int n = tile * 16 + m;
    int y = wy * stl + n / wsq, x = wx * stl + n % wsq;
    size_t p = (size_t)(b * PL) + y * HL + x;
    float val = acc[r] * ww * zbuf[m][head];
    __hip_atomic_fetch_add(&attn_acc[p * CB + c], val,
                           __ATOMIC_RELAXED, __HIP_MEMORY_SCOPE_AGENT);
  }
}

// ---------------- kernel 5: out = (fold_acc @ wo^T + cov*bo) / 2 ----------------
__global__ __launch_bounds__(256) void k_out(const float* attn_acc, const float* bo,
                                             unsigned char* wsb, float* outp) {
  __shared__ unsigned short abuf[16][128];
  const unsigned short* woF = (const unsigned short*)(wsb + OFF_WOF);
  const int tid = threadIdx.x, lane = tid & 31, wave = tid >> 5;
  const int half = lane >> 4, ml = lane & 15;
  const int pix0 = blockIdx.x * 16, b = pix0 / PL, p0 = pix0 % PL;

  for (int s = tid; s < 512; s += 256) {     // 16B f32 loads, 8B bf16 stores
    int m = s >> 5, c0 = (s & 31) * 4;
    float4 f = *(const float4*)&attn_acc[(size_t)(b * PL + p0 + m) * CB + c0];
    ushort4 t;
    t.x = f2bf(f.x); t.y = f2bf(f.y); t.z = f2bf(f.z); t.w = f2bf(f.w);
    *(ushort4*)&abuf[m][c0] = t;
  }
  __syncthreads();

  v8f acc = {0.f, 0.f, 0.f, 0.f, 0.f, 0.f, 0.f, 0.f};
  for (int kc = 0; kc < 4; ++kc) {
    v16u16 a = loadA(&abuf[ml][kc * 32], half);
    v16u16 bb = *(const v16u16*)(woF + ((kc * 8 + wave) * 32 + lane) * 16);
    acc = wmma_bf16(a, bb, acc);
  }
  const int c = wave * 16 + ml;
  const float bov = bo[c];
#pragma unroll
  for (int r = 0; r < 8; ++r) {
    int m = half * 8 + r;
    int p = p0 + m;
    int y = p >> 7, x = p & 127;
    int cov = cov1d(y, 16, 8, NH1) * cov1d(x, 16, 8, NH1)
            + cov1d(y, 32, 16, NH2) * cov1d(x, 32, 16, NH2);
    outp[(size_t)(b * CB + c) * PL + p] = 0.5f * (acc[r] + (float)cov * bov);
  }
}

// ---------------- kernel 6: final = alpha*out + (1-alpha)*low ----------------
__global__ __launch_bounds__(256) void k_final(const float* outp, const float* low,
                                               const float* alpha, float* finp, int n) {
  int i = blockIdx.x * 256 + threadIdx.x;
  if (i < n) {
    float a = alpha[0];
    finp[i] = a * outp[i] + (1.f - a) * low[i];
  }
}

extern "C" void kernel_launch(void* const* d_in, const int* in_sizes, int n_in,
                              void* d_out, int out_size, void* d_ws, size_t ws_size,
                              hipStream_t stream) {
  const float* high   = (const float*)d_in[0];
  const float* low    = (const float*)d_in[1];
  const float* conv_w = (const float*)d_in[2];
  const float* conv_b = (const float*)d_in[3];
  const float* ln_g   = (const float*)d_in[4];
  const float* ln_b   = (const float*)d_in[5];
  const float* wq = (const float*)d_in[6];
  const float* bq = (const float*)d_in[7];
  const float* wk = (const float*)d_in[8];
  const float* bk = (const float*)d_in[9];
  const float* wv = (const float*)d_in[10];
  const float* bv = (const float*)d_in[11];
  const float* wo = (const float*)d_in[12];
  const float* bo = (const float*)d_in[13];
  const float* wws   = (const float*)d_in[14];
  const float* alpha = (const float*)d_in[15];

  unsigned char* wsb = (unsigned char*)d_ws;
  float* outp = (float*)d_out;
  const int NTOT = BN * CB * PL;            // 16777216 per output tensor
  float* attn = outp + (size_t)NTOT;        // second half doubles as fold accumulator

  k_prep<<<128, 256, 0, stream>>>(conv_w, wq, wk, wv, wo, wsb);
  k_zero<<<(NTOT + 255) / 256, 256, 0, stream>>>(attn, NTOT);
  k_hs<<<BN * PH / 16, 256, 0, stream>>>(high, conv_b, ln_g, ln_b, bk, bv, wsb);
  k_q<<<BN * PL / 16, 256, 0, stream>>>(low, ln_g, ln_b, bq, wsb);
  k_kv<<<BN * NH1 * NH1, 256, 0, stream>>>(wsb, 8, NH1, OFF_KV8, OFF_KM8);
  k_kv<<<BN * NH2 * NH2, 256, 0, stream>>>(wsb, 16, NH2, OFF_KV16, OFF_KM16);
  k_att<<<BN * NH1 * NH1 * 16, 256, 0, stream>>>(wsb, wws, 0, attn, 8, NH1, 16, OFF_KV8, OFF_KM8);
  k_att<<<BN * NH2 * NH2 * 64, 256, 0, stream>>>(wsb, wws, 1, attn, 16, NH2, 64, OFF_KV16, OFF_KM16);
  k_out<<<BN * PL / 16, 256, 0, stream>>>(attn, bo, wsb, outp);
  k_final<<<(NTOT + 255) / 256, 256, 0, stream>>>(outp, low, alpha, attn, NTOT);
}
